// HungarianMatcher_17300128268908
// MI455X (gfx1250) — matrix operand order repaired and checked by reading, not compiled, not run on hardware
//
#include <hip/hip_runtime.h>

typedef __attribute__((ext_vector_type(16))) _Float16 v16h;
typedef __attribute__((ext_vector_type(8)))  float    v8f;

#define BS_    16
#define NQ_    900
#define NC_    256
#define NT_    800
#define NQTOT  (BS_*NQ_)        // 14400 query rows
#define QTILES (NQTOT/16)       // 900
#define TTILES (NT_/16)         // 50
#define NTILES (QTILES*TTILES)  // 45000
#define FRAG_PER_TILE (8*32)    // 8 K-steps x 32 lanes, each one v16h

// ---------------------------------------------------------------------------
// Kernel 1: per-box derived values (xyxy + area) for pred and tgt boxes.
// ---------------------------------------------------------------------------
__global__ void box_prep(const float* __restrict__ pb, const float* __restrict__ tb,
                         float4* __restrict__ qxy, float* __restrict__ qar,
                         float4* __restrict__ txy, float* __restrict__ tar) {
    int i = blockIdx.x * blockDim.x + threadIdx.x;
    if (i < NQTOT) {
        float4 b = ((const float4*)pb)[i];       // cx, cy, w, h
        float x0 = b.x - 0.5f * b.z, y0 = b.y - 0.5f * b.w;
        float x1 = b.x + 0.5f * b.z, y1 = b.y + 0.5f * b.w;
        qxy[i] = make_float4(x0, y0, x1, y1);
        qar[i] = (x1 - x0) * (y1 - y0);
    } else if (i < NQTOT + NT_) {
        int j = i - NQTOT;
        float4 b = ((const float4*)tb)[j];
        float x0 = b.x - 0.5f * b.z, y0 = b.y - 0.5f * b.w;
        float x1 = b.x + 0.5f * b.z, y1 = b.y + 0.5f * b.w;
        txy[j] = make_float4(x0, y0, x1, y1);
        tar[j] = (x1 - x0) * (y1 - y0);
    }
}

// ---------------------------------------------------------------------------
// Kernel 2: one-hot B fragments in WMMA B-matrix (32x16 f16) register layout.
// Lane L: column N = L&15; K = 32*s + (L>=16 ? 8 : 0) + (e&7) + ((e&8) ? 16 : 0), e = 0..15.
// Grid: TTILES blocks x 256 threads (thread t -> step s = t>>5, lane = t&31).
// ---------------------------------------------------------------------------
__global__ void onehot_b(const int* __restrict__ labels, _Float16* __restrict__ Bswz) {
    int tt   = blockIdx.x;
    int t    = threadIdx.x;
    int s    = t >> 5;
    int lane = t & 31;
    int n    = lane & 15;
    int kbase = s * 32 + ((lane >> 4) << 3);
    int lbl = labels[tt * 16 + n];
    v16h f;
#pragma unroll
    for (int e = 0; e < 16; e++)
        f[e] = (lbl == kbase + (e & 7) + ((e & 8) << 1)) ? (_Float16)1.0f : (_Float16)0.0f;
    *((v16h*)(Bswz + ((size_t)(tt * 8 + s) * 32 + lane) * 16)) = f;
}

// ---------------------------------------------------------------------------
// Kernel 3: row softmax (stable) over 256 classes, 16 rows per block,
// emit f16 probs pre-swizzled into the 16-bit A-matrix 16x32 fragment layout:
//   for class k within a 32-wide step: lane = M + ((k&8)?16:0),
//   element e = (k&7) + ((k&16)?8:0).
// ---------------------------------------------------------------------------
__global__ __launch_bounds__(256) void softmax_swz(const float* __restrict__ logits,
                                                   _Float16* __restrict__ Aswz) {
    __shared__ float red[16][17];
    __shared__ __align__(16) _Float16 atile[FRAG_PER_TILE * 16];  // 8 KB
    int qt = blockIdx.x;
    int t  = threadIdx.x;
    int r  = t >> 4;       // row within tile
    int i  = t & 15;       // 16 threads per row
    const float* row = logits + (size_t)(qt * 16 + r) * NC_;

    float v[16];
#pragma unroll
    for (int j = 0; j < 16; j++) v[j] = row[i + 16 * j];

    float lm = v[0];
#pragma unroll
    for (int j = 1; j < 16; j++) lm = fmaxf(lm, v[j]);
    red[r][i] = lm;
    __syncthreads();
    float rm = red[r][0];
#pragma unroll
    for (int k = 1; k < 16; k++) rm = fmaxf(rm, red[r][k]);

    float s = 0.0f;
#pragma unroll
    for (int j = 0; j < 16; j++) { v[j] = __expf(v[j] - rm); s += v[j]; }
    __syncthreads();
    red[r][i] = s;
    __syncthreads();
    float rs = 0.0f;
#pragma unroll
    for (int k = 0; k < 16; k++) rs += red[r][k];
    float inv = __builtin_amdgcn_rcpf(rs);

#pragma unroll
    for (int j = 0; j < 16; j++) {
        int c    = i + 16 * j;            // class index (== K)
        int st   = c >> 5;                // K-step
        int k32  = c & 31;
        int lane = r + ((k32 & 8) ? 16 : 0);
        int e    = (k32 & 7) + ((k32 & 16) ? 8 : 0);
        atile[(st * 32 + lane) * 16 + e] = (_Float16)(v[j] * inv);
    }
    __syncthreads();

    // coalesced 8KB copy to global fragment buffer (256 threads x 32B)
    const uint4* src = (const uint4*)atile;
    uint4* dst = (uint4*)(Aswz + (size_t)qt * FRAG_PER_TILE * 16);
    dst[t]       = src[t];
    dst[t + 256] = src[t + 256];
}

// ---------------------------------------------------------------------------
// Kernel 4: fused cost tiles. One wave per 16x16 output tile.
//   acc = P_tile(16x256) @ OneHot(256x16) via 8x v_wmma_f32_16x16x32_f16,
//   then per-lane epilogue: C = 5*L1 - acc - 2*GIoU.
// D layout: lane holds (M = i + (lane>=16 ? 8 : 0), N = lane&15) for acc[i].
// Grid: NTILES/8 blocks x 256 threads (8 waves); consecutive waves share qt.
// ---------------------------------------------------------------------------
__global__ __launch_bounds__(256) void cost_kernel(
    const _Float16* __restrict__ Aswz, const _Float16* __restrict__ Bswz,
    const float* __restrict__ pb, const float* __restrict__ tb,
    const float4* __restrict__ qxy, const float* __restrict__ qar,
    const float4* __restrict__ txy, const float* __restrict__ tar,
    float* __restrict__ out) {
    int w    = blockIdx.x * 8 + (threadIdx.x >> 5);     // tile id, < NTILES
    int lane = threadIdx.x & 31;
    int qt   = w / TTILES;
    int tt   = w - qt * TTILES;

    const v16h* Ap = (const v16h*)Aswz + (size_t)qt * FRAG_PER_TILE + lane;
    const v16h* Bp = (const v16h*)Bswz + (size_t)tt * FRAG_PER_TILE + lane;

    v8f acc = {};
#pragma unroll
    for (int s = 0; s < 8; s++) {
        v16h a = Ap[s * 32];
        v16h b = Bp[s * 32];
        acc = __builtin_amdgcn_wmma_f32_16x16x32_f16(
                  false, a, false, b, (short)0, acc, false, false);
    }

    // ---- epilogue: L1 + GIoU for this lane's 8 C-elements ----
    int n = tt * 16 + (lane & 15);
    float4 t2 = ((const float4*)tb)[n];   // tgt cxcywh (broadcast across i)
    float4 x2 = txy[n];
    float  a2 = tar[n];
    int mbase = qt * 16 + ((lane >> 4) << 3);

#pragma unroll
    for (int i = 0; i < 8; i++) {
        int m = mbase + i;
        float4 t1 = ((const float4*)pb)[m];   // same for all 16 lanes of a half
        float4 x1 = qxy[m];
        float  a1 = qar[m];

        float l1 = fabsf(t1.x - t2.x) + fabsf(t1.y - t2.y) +
                   fabsf(t1.z - t2.z) + fabsf(t1.w - t2.w);

        float ix0 = fmaxf(x1.x, x2.x), iy0 = fmaxf(x1.y, x2.y);
        float ix1 = fminf(x1.z, x2.z), iy1 = fminf(x1.w, x2.w);
        float iw  = fmaxf(ix1 - ix0, 0.0f), ih = fmaxf(iy1 - iy0, 0.0f);
        float inter = iw * ih;
        float uni   = a1 + a2 - inter;
        float iou   = inter * __builtin_amdgcn_rcpf(uni + 1e-7f);

        float cx0 = fminf(x1.x, x2.x), cy0 = fminf(x1.y, x2.y);
        float cx1 = fmaxf(x1.z, x2.z), cy1 = fmaxf(x1.w, x2.w);
        float cw  = fmaxf(cx1 - cx0, 0.0f), ch = fmaxf(cy1 - cy0, 0.0f);
        float car = cw * ch;
        float giou = iou - (car - uni) * __builtin_amdgcn_rcpf(car + 1e-7f);

        // C = 5*L1 + 1*(-prob) + 2*(-giou)
        out[(size_t)m * NT_ + n] = 5.0f * l1 - acc[i] - 2.0f * giou;
    }
}

// ---------------------------------------------------------------------------
extern "C" void kernel_launch(void* const* d_in, const int* in_sizes, int n_in,
                              void* d_out, int out_size, void* d_ws, size_t ws_size,
                              hipStream_t stream) {
    const float* pred_logits = (const float*)d_in[0];   // [16,900,256] f32
    const float* pred_boxes  = (const float*)d_in[1];   // [14400,4]    f32
    const int*   tgt_labels  = (const int*)  d_in[2];   // [800]        int
    const float* tgt_boxes   = (const float*)d_in[3];   // [800,4]      f32
    float* out = (float*)d_out;                         // [16,900,800] f32

    // workspace carve-up (all 256B-aligned)
    char* ws = (char*)d_ws;
    size_t oA   = 0;                                  // A frags: 14400*256 f16
    size_t oB   = oA + (size_t)NQTOT * NC_ * 2;       // 7,372,800
    size_t oQXY = oB + (size_t)TTILES * FRAG_PER_TILE * 16 * 2;  // +409,600
    size_t oQAR = oQXY + (size_t)NQTOT * sizeof(float4);
    size_t oTXY = oQAR + (((size_t)NQTOT * sizeof(float) + 255) & ~(size_t)255);
    size_t oTAR = oTXY + (size_t)NT_ * sizeof(float4);

    _Float16* Aswz = (_Float16*)(ws + oA);
    _Float16* Bswz = (_Float16*)(ws + oB);
    float4*   qxy  = (float4*)(ws + oQXY);
    float*    qar  = (float*) (ws + oQAR);
    float4*   txy  = (float4*)(ws + oTXY);
    float*    tar  = (float*) (ws + oTAR);

    box_prep<<<(NQTOT + NT_ + 255) / 256, 256, 0, stream>>>(
        pred_boxes, tgt_boxes, qxy, qar, txy, tar);

    onehot_b<<<TTILES, 256, 0, stream>>>(tgt_labels, Bswz);

    softmax_swz<<<QTILES, 256, 0, stream>>>(pred_logits, Aswz);

    cost_kernel<<<NTILES / 8, 256, 0, stream>>>(
        Aswz, Bswz, pred_boxes, tgt_boxes, qxy, qar, txy, tar, out);
}
